// GraphicalStructureModeling_47708496724085
// MI455X (gfx1250) — matrix operand order repaired and checked by reading, not compiled, not run on hardware
//
#include <hip/hip_runtime.h>
#include <hip/hip_bf16.h>
#include <math.h>

typedef __attribute__((ext_vector_type(16))) __bf16 v16bf;
typedef __attribute__((ext_vector_type(8)))  float  v8f;
typedef int v2i_t __attribute__((vector_size(8)));   // b64 payload type

#define NN 256

// ---------------------------------------------------------------------------
// CDNA5 async global->LDS path (ASYNCcnt).  Gated so the file still compiles
// (with synchronous staging) if the toolchain lacks the builtins.
// ---------------------------------------------------------------------------
#if defined(__gfx1250__) && __has_builtin(__builtin_amdgcn_global_load_async_to_lds_b64) && __has_builtin(__builtin_amdgcn_s_wait_asynccnt)
  #define ATHENA_ASYNC_LDS 1
#endif

#ifdef ATHENA_ASYNC_LDS
#pragma message("ATHENA: gfx1250 async global->LDS path ENABLED")
__device__ __forceinline__ void tile_async_b64(float* lds_dst, const float* gsrc)
{
    __builtin_amdgcn_global_load_async_to_lds_b64(
        (__attribute__((address_space(1))) v2i_t*)gsrc,
        (__attribute__((address_space(3))) v2i_t*)lds_dst,
        /*offset=*/0, /*cpol=*/0);
}
__device__ __forceinline__ void wait_async0() { __builtin_amdgcn_s_wait_asynccnt(0); }
#else
#pragma message("ATHENA: gfx1250 async global->LDS path DISABLED (builtin missing)")
#endif

// ---------------------------------------------------------------------------
// Pack a K x 256 f32 matrix into bf16 WMMA B-fragment order:
//   Bp[(((kb*16 + ntile)*32 + lane)*16 + idx)] = B[kb*32 + (lane/16)*16 + idx][ntile*16 + lane%16]
// so a wave's B fragment is one contiguous 32-byte per-lane load.
// grid.x = K, block = 256  (one bf16 element per thread).
// ---------------------------------------------------------------------------
__global__ __launch_bounds__(256) void pack_b_kernel(
    const float* __restrict__ B, __bf16* __restrict__ Bp)
{
    const int o    = blockIdx.x * 256 + threadIdx.x;
    const int tile = o >> 9;          // kb*16 + ntile
    const int kb   = tile >> 4;
    const int nt   = tile & 15;
    const int lane = (o >> 4) & 31;
    const int idx  = o & 15;
    const int kk   = ((lane >> 4) << 4) + idx;
    const int col  = (nt << 4) + (lane & 15);
    Bp[o] = (__bf16)B[(size_t)(kb * 32 + kk) * NN + col];
}

// ---------------------------------------------------------------------------
// GEMM: C[M x 256] = A[M x K] (f32) @ B (pre-packed bf16), f32 accum.
// grid.x = M/16, block = 256 (8 waves); wave w owns columns [32w,32w+32).
// A K-tiles double-buffered in LDS via async DMA.
// ---------------------------------------------------------------------------
__global__ __launch_bounds__(256) void gemm_bf16_wmma(
    const float* __restrict__ A, const __bf16* __restrict__ Bp,
    float* __restrict__ C, int K)
{
    __shared__ float Atile[2][16][32];
    const int tid  = threadIdx.x;
    const int lane = tid & 31;
    const int wave = tid >> 5;
    const int m0   = blockIdx.x << 4;
    const int mrow = lane & 15;
    const int half = lane >> 4;
    const int arow = tid >> 4;            // staging: row 0..15
    const int acol = (tid & 15) << 1;     // staging: col pair
    const v16bf* Bv = (const v16bf*)Bp;

    v8f acc0 = {}, acc1 = {};
    int buf = 0;

#ifdef ATHENA_ASYNC_LDS
    tile_async_b64(&Atile[0][arow][acol], A + (size_t)(m0 + arow) * K + acol);
#endif

    for (int kb = 0; kb < K; kb += 32) {
#ifdef ATHENA_ASYNC_LDS
        wait_async0();                    // my slice of tile[buf] landed
        __syncthreads();                  // everyone's slice landed
        if (kb + 32 < K)                  // overlap next DMA with compute
            tile_async_b64(&Atile[buf ^ 1][arow][acol],
                           A + (size_t)(m0 + arow) * K + kb + 32 + acol);
#else
        __syncthreads();
        if (kb + 32 < K)
            __builtin_prefetch(&A[(size_t)(m0 + arow) * K + kb + 32 + acol], 0, 1);
        Atile[buf][arow][acol]     = A[(size_t)(m0 + arow) * K + kb + acol];
        Atile[buf][arow][acol + 1] = A[(size_t)(m0 + arow) * K + kb + acol + 1];
        __syncthreads();
#endif
        // A fragment: contiguous k = 16*half + s (same order as packed B), f32->bf16 convert in-reg.
        const float* Ar = &Atile[buf][mrow][0];
        v16bf a;
        #pragma unroll
        for (int vg = 0; vg < 8; ++vg) {
            #pragma unroll
            for (int p = 0; p < 2; ++p) {
                int k = (half << 4) + (vg << 1) + p;
                a[2 * vg + p] = (__bf16)Ar[k];
            }
        }

        // B fragments: one 32B contiguous per-lane load each (2x b128).
        size_t tbase = ((size_t)(kb >> 5) * 16 + (wave << 1)) * 32 + lane;
        v16bf b0 = Bv[tbase];
        v16bf b1 = Bv[tbase + 32];

        acc0 = __builtin_amdgcn_wmma_f32_16x16x32_bf16(false, a, false, b0, (short)0, acc0, false, false);
        acc1 = __builtin_amdgcn_wmma_f32_16x16x32_bf16(false, a, false, b1, (short)0, acc1, false, false);
#ifdef ATHENA_ASYNC_LDS
        buf ^= 1;
#endif
    }

    const int n0 = wave << 5;
    #pragma unroll
    for (int r = 0; r < 8; ++r) {
        int m = m0 + r + (half << 3);
        C[(size_t)m * NN + n0 + mrow]      = acc0[r];
        C[(size_t)m * NN + n0 + 16 + mrow] = acc1[r];
    }
}

// ---------------------------------------------------------------------------
// Row sum-of-squares of h: sq[row] = sum_c h[row][c]^2.  Wave per row.
// ---------------------------------------------------------------------------
__global__ __launch_bounds__(256) void sumsq_kernel(
    const float* __restrict__ h, float* __restrict__ sq)
{
    const int lane = threadIdx.x & 31;
    const int wave = threadIdx.x >> 5;
    const int row  = blockIdx.x * 8 + wave;
    const float* p = h + (size_t)row * NN;
    float a = 0.0f;
    #pragma unroll
    for (int c = lane; c < NN; c += 32) { float v = p[c]; a += v * v; }
    #pragma unroll
    for (int off = 16; off > 0; off >>= 1) a += __shfl_down(a, off, 32);
    if (lane == 0) sq[row] = a;
}

// ---------------------------------------------------------------------------
// Fused Gram + distance: per batch b, G = H H^T via bf16 WMMA,
// Dis = sqrt(relu(sq_i + sq_j - 2G)).  grid = (16 row-tiles, 128 batches).
// B side comes from Hp (h packed flat as one K=32768 B-matrix: per-batch
// 256-row blocks align exactly with 32-row K-tiles).
// ---------------------------------------------------------------------------
__global__ __launch_bounds__(256) void dist_wmma(
    const float* __restrict__ h, const __bf16* __restrict__ Hp,
    const float* __restrict__ sq, float* __restrict__ Dis)
{
    __shared__ float Atile[2][16][32];
    const int b    = blockIdx.y;
    const int m0   = blockIdx.x << 4;
    const int tid  = threadIdx.x;
    const int lane = tid & 31;
    const int wave = tid >> 5;
    const int mrow = lane & 15;
    const int half = lane >> 4;
    const int arow = tid >> 4;
    const int acol = (tid & 15) << 1;
    const float* H = h + (size_t)b * NN * NN;
    const v16bf* Bv = (const v16bf*)Hp;

    v8f acc0 = {}, acc1 = {};
    int buf = 0;

#ifdef ATHENA_ASYNC_LDS
    tile_async_b64(&Atile[0][arow][acol], H + (size_t)(m0 + arow) * NN + acol);
#endif

    for (int kb = 0; kb < NN; kb += 32) {
#ifdef ATHENA_ASYNC_LDS
        wait_async0();
        __syncthreads();
        if (kb + 32 < NN)
            tile_async_b64(&Atile[buf ^ 1][arow][acol],
                           H + (size_t)(m0 + arow) * NN + kb + 32 + acol);
#else
        __syncthreads();
        Atile[buf][arow][acol]     = H[(size_t)(m0 + arow) * NN + kb + acol];
        Atile[buf][arow][acol + 1] = H[(size_t)(m0 + arow) * NN + kb + acol + 1];
        __syncthreads();
#endif
        const float* Ar = &Atile[buf][mrow][0];
        v16bf a;
        #pragma unroll
        for (int vg = 0; vg < 8; ++vg) {
            #pragma unroll
            for (int p = 0; p < 2; ++p) {
                int k = (half << 4) + (vg << 1) + p;
                a[2 * vg + p] = (__bf16)Ar[k];
            }
        }

        // global K-row = b*256 + kb  ->  tile base (b*8 + kb/32)*16 + ntile
        size_t tbase = ((size_t)(b * 8 + (kb >> 5)) * 16 + (wave << 1)) * 32 + lane;
        v16bf b0 = Bv[tbase];
        v16bf b1 = Bv[tbase + 32];

        acc0 = __builtin_amdgcn_wmma_f32_16x16x32_bf16(false, a, false, b0, (short)0, acc0, false, false);
        acc1 = __builtin_amdgcn_wmma_f32_16x16x32_bf16(false, a, false, b1, (short)0, acc1, false, false);
#ifdef ATHENA_ASYNC_LDS
        buf ^= 1;
#endif
    }

    const int n0 = wave << 5;
    const float* SQ = sq + (size_t)b * NN;
    float* Db = Dis + (size_t)b * NN * NN;
    #pragma unroll
    for (int r = 0; r < 8; ++r) {
        int m = m0 + r + (half << 3);
        float sm = SQ[m];
        {
            int n = n0 + mrow;
            float d = sm + SQ[n] - 2.0f * acc0[r];
            Db[(size_t)m * NN + n] = sqrtf(fmaxf(d, 0.0f));
        }
        {
            int n = n0 + 16 + mrow;
            float d = sm + SQ[n] - 2.0f * acc1[r];
            Db[(size_t)m * NN + n] = sqrtf(fmaxf(d, 0.0f));
        }
    }
}

// ---------------------------------------------------------------------------
// Edge-building loop: one block per batch element, 254 sequential steps.
// Incremental LDS state: deg[], S_non[i] = sum_{j non-iso} Dis[i,j], and
// num_noniso; only the step's argmin node can flip iso -> non-iso, so each
// step is O(N) + one 1KB L2-hot Dis row read.
// ---------------------------------------------------------------------------
__global__ __launch_bounds__(256) void build_edge_kernel(
    const float* __restrict__ Dis, float* __restrict__ At)
{
    const float FINF = __builtin_inff();
    const int b = blockIdx.x;
    const int t = threadIdx.x;
    const float* D = Dis + (size_t)b * NN * NN;
    float* A = At + (size_t)b * NN * NN;

    __shared__ float s_rowT[NN];
    __shared__ float s_Snon[NN];
    __shared__ int   s_deg[NN];
    __shared__ float r_v1[NN];
    __shared__ float r_v2[NN];
    __shared__ int   r_i1[NN];
    __shared__ int   s_nn;
    __shared__ int   s_addcnt;
    __shared__ int   s_minidx;
    __shared__ float s_sv1, s_soloth, s_cluth;

    for (int j = t; j < NN * NN; j += NN) A[j] = 0.0f;

    // rowT via column sums (Dis symmetric) -> coalesced.
    float acc = 0.0f;
    for (int j = 0; j < NN; ++j) acc += D[(size_t)j * NN + t];
    s_rowT[t] = acc;

    // Init: argmin over off-diagonal, ties -> lowest flat index.
    float mv = FINF; int mi = 0;
    for (int j = 0; j < NN; ++j) {
        if (j == t) continue;
        float v = D[(size_t)t * NN + j];
        if (v < mv) { mv = v; mi = t * NN + j; }
    }
    r_v1[t] = mv; r_i1[t] = mi;
    __syncthreads();
    for (int s = NN / 2; s > 0; s >>= 1) {
        if (t < s) {
            float v2 = r_v1[t + s]; int i2 = r_i1[t + s];
            if (v2 < r_v1[t] || (v2 == r_v1[t] && i2 < r_i1[t])) { r_v1[t] = v2; r_i1[t] = i2; }
        }
        __syncthreads();
    }
    const int r0 = r_i1[0] / NN;
    const int c0 = r_i1[0] % NN;
    if (t == 0) { A[(size_t)r0 * NN + c0] = 1.0f; s_nn = 2; }
    s_deg[t]  = (t == r0 || t == c0) ? 1 : 0;
    s_Snon[t] = D[(size_t)r0 * NN + t] + D[(size_t)c0 * NN + t];
    __syncthreads();

    for (int step = 0; step < NN - 2; ++step) {
        const bool last = (step == NN - 3);
        const int  nn   = s_nn;
        const int  niso = NN - nn;
        const bool iso  = (s_deg[t] == 0);

        float sm = (iso && nn > 0) ? (s_Snon[t] / (float)nn) : FINF;
        r_v1[t] = sm; r_v2[t] = FINF; r_i1[t] = t;
        __syncthreads();
        for (int s = NN / 2; s > 0; s >>= 1) {
            if (t < s) {
                float av1 = r_v1[t], av2 = r_v2[t]; int ai = r_i1[t];
                float bv1 = r_v1[t + s], bv2 = r_v2[t + s]; int bi = r_i1[t + s];
                if (bv1 < av1 || (bv1 == av1 && bi < ai)) {
                    r_v1[t] = bv1; r_i1[t] = bi; r_v2[t] = fminf(av1, bv2);
                } else {
                    r_v2[t] = fminf(bv1, av2);
                }
            }
            __syncthreads();
        }
        if (t == 0) {
            s_minidx = r_i1[0];
            s_sv1    = r_v1[0];
            s_soloth = last ? r_v1[0] : r_v2[0];
        }
        __syncthreads();

        float cm = (!iso && niso > 0) ? ((s_rowT[t] - s_Snon[t]) / (float)niso) : FINF;
        r_v1[t] = cm; r_v2[t] = FINF; r_i1[t] = t;
        __syncthreads();
        for (int s = NN / 2; s > 0; s >>= 1) {
            if (t < s) {
                float av1 = r_v1[t], av2 = r_v2[t]; int ai = r_i1[t];
                float bv1 = r_v1[t + s], bv2 = r_v2[t + s]; int bi = r_i1[t + s];
                if (bv1 < av1 || (bv1 == av1 && bi < ai)) {
                    r_v1[t] = bv1; r_i1[t] = bi; r_v2[t] = fminf(av1, bv2);
                } else {
                    r_v2[t] = fminf(bv1, av2);
                }
            }
            __syncthreads();
        }
        if (t == 0) { s_cluth = r_v2[0]; s_addcnt = 0; }
        __syncthreads();

        const int   minidx = s_minidx;
        const float sv1    = s_sv1;
        const float soloth = s_soloth;
        const float cluth  = s_cluth;

        const float rowv = D[(size_t)minidx * NN + t];
        int added = 0;
        if (sv1 < FINF && s_deg[t] > 0) {
            if (soloth < FINF && rowv < soloth) { A[(size_t)minidx * NN + t] += 1.0f; ++added; }
            if (cluth  < FINF && rowv < cluth)  { A[(size_t)t * NN + minidx] += 1.0f; ++added; }
        }
        if (added) {
            s_deg[t] += added;
            atomicAdd(&s_addcnt, added);
        }
        __syncthreads();
        if (s_addcnt > 0) {
            s_Snon[t] += rowv;
            if (t == 0) { s_deg[minidx] += s_addcnt; s_nn += 1; }
        }
        __syncthreads();
    }
}

// ---------------------------------------------------------------------------
extern "C" void kernel_launch(void* const* d_in, const int* in_sizes, int n_in,
                              void* d_out, int out_size, void* d_ws, size_t ws_size,
                              hipStream_t stream)
{
    (void)in_sizes; (void)n_in; (void)out_size; (void)ws_size;
    const float* x  = (const float*)d_in[0];   // (128, 256, 2048)
    const float* W1 = (const float*)d_in[1];   // (256, 256)
    const float* W2 = (const float*)d_in[2];   // (2048, 256)
    float* At = (float*)d_out;                 // (128, 256, 256)

    const size_t M = 128 * 256;                // 32768 rows
    float*  t   = (float*)d_ws;                // M x 256
    float*  h   = t + M * NN;                  // M x 256
    float*  Dis = h + M * NN;                  // 128 x 256 x 256
    float*  sq  = Dis + (size_t)128 * NN * NN; // M
    __bf16* W2p = (__bf16*)(sq + M);           // 2048*256 packed bf16
    __bf16* W1p = W2p + (size_t)2048 * NN;     // 256*256
    __bf16* Hp  = W1p + (size_t)NN * NN;       // M*256

    pack_b_kernel<<<2048, 256, 0, stream>>>(W2, W2p);
    pack_b_kernel<<<256, 256, 0, stream>>>(W1, W1p);
    gemm_bf16_wmma<<<(int)(M / 16), 256, 0, stream>>>(x, W2p, t, 2048);
    gemm_bf16_wmma<<<(int)(M / 16), 256, 0, stream>>>(t, W1p, h, 256);
    sumsq_kernel<<<(int)(M / 8), 256, 0, stream>>>(h, sq);
    pack_b_kernel<<<(int)M, 256, 0, stream>>>(h, Hp);
    dim3 dg(NN / 16, 128);
    dist_wmma<<<dg, 256, 0, stream>>>(h, Hp, sq, Dis);
    build_edge_kernel<<<128, 256, 0, stream>>>(Dis, At);
}
